// WOQ_Linear_Add_Add_Model_82841329206005
// MI455X (gfx1250) — compile-verified, matrix operand order checked by reading, not run on hardware
//
#include <hip/hip_runtime.h>
#include <hip/hip_bf16.h>

typedef __attribute__((ext_vector_type(16))) __bf16 v16bf;
typedef __attribute__((ext_vector_type(8)))  __bf16 v8bf;
typedef __attribute__((ext_vector_type(8)))  float  v8f;

constexpr int NROWS = 16384;   // batch
constexpr int KDIM  = 4096;    // IN
constexpr int ODIM  = 4096;    // OUT
constexpr int BM = 256;        // block tile M (4 waves x 64)
constexpr int BN = 128;        // block tile N (2 waves x 64)
constexpr int BK = 32;         // K step (wmma 16x16x32)
constexpr int XS = 40;         // LDS row stride (elems): 80B, 16 rows -> 16 distinct banks
constexpr int WS = 40;

__global__ __launch_bounds__(256)
void woq_linear_add_add_kernel(const __bf16* __restrict__ xg,
                               const int*    __restrict__ qw,
                               const __bf16* __restrict__ scalesp,
                               const __bf16* __restrict__ biasp,
                               const __bf16* __restrict__ add1p,
                               const __bf16* __restrict__ add2p,
                               __bf16*       __restrict__ outp)
{
    __shared__ __align__(16) __bf16 sX[BM * XS];   // 20.0 KB
    __shared__ __align__(16) __bf16 sW[BN * WS];   // 10.0 KB

    const int tid   = threadIdx.x;
    const int lane  = tid & 31;
    const int wid   = tid >> 5;     // 0..7
    const int waveM = wid & 3;      // 0..3  -> 4 * 64 rows = 256
    const int waveN = wid >> 2;     // 0..1  -> 2 * 64 cols = 128

    const int rowBase = blockIdx.y * BM;
    const int colBase = blockIdx.x * BN;

    const int l15 = lane & 15;
    const int kb  = (lane >> 4) * 8;   // K sub-offset per ISA 16-bit A/B fragment layout

    v8f acc[4][4] = {};

    for (int kt = 0; kt < KDIM; kt += BK) {
        // ---- stage x tile: BM x BK bf16 = 16KB -> 1024 x 16B chunks, 4 per thread
        #pragma unroll
        for (int j = 0; j < 4; ++j) {
            int c  = tid + j * 256;
            int r  = c >> 2;            // 4 chunks per 64B row
            int ck = (c & 3) * 8;
            const uint4* src = (const uint4*)(xg + (size_t)(rowBase + r) * KDIM + kt + ck);
            *(uint4*)&sX[r * XS + ck] = *src;
        }
        // ---- stage + dequant weight tile: BN x BK int4-codes -> bf16 (q-8)
        // scale is per-output-channel -> folded into epilogue, exact small ints here
        #pragma unroll
        for (int j = 0; j < 2; ++j) {
            int c  = tid + j * 256;
            int r  = c >> 2;            // 0..127
            int ck = (c & 3) * 8;
            const int4* qp = (const int4*)(qw + (size_t)(colBase + r) * KDIM + kt + ck);
            int4 q0 = qp[0];
            int4 q1 = qp[1];
            v8bf wv;
            wv[0] = (__bf16)(float)(q0.x - 8);
            wv[1] = (__bf16)(float)(q0.y - 8);
            wv[2] = (__bf16)(float)(q0.z - 8);
            wv[3] = (__bf16)(float)(q0.w - 8);
            wv[4] = (__bf16)(float)(q1.x - 8);
            wv[5] = (__bf16)(float)(q1.y - 8);
            wv[6] = (__bf16)(float)(q1.z - 8);
            wv[7] = (__bf16)(float)(q1.w - 8);
            *(v8bf*)&sW[r * WS + ck] = wv;
        }

        // ---- prefetch next K tile into cache hierarchy while this one computes
        if (kt + BK < KDIM) {
            __builtin_prefetch(xg + (size_t)(rowBase + (tid >> 2) * 2) * KDIM + kt + BK, 0, 1);
            __builtin_prefetch(qw + (size_t)(colBase + (tid >> 1)) * KDIM + kt + BK, 0, 1);
        }

        __syncthreads();

        // ---- fragment loads from LDS (ds_load_b128 x2 per fragment)
        v16bf a[4], b[4];
        #pragma unroll
        for (int f = 0; f < 4; ++f) {
            int row = waveM * 64 + f * 16 + l15;
            v8bf lo = *(const v8bf*)&sX[row * XS + kb];
            v8bf hi = *(const v8bf*)&sX[row * XS + 16 + kb];
            #pragma unroll
            for (int i = 0; i < 8; ++i) { a[f][i] = lo[i]; a[f][8 + i] = hi[i]; }
        }
        #pragma unroll
        for (int f = 0; f < 4; ++f) {
            int col = waveN * 64 + f * 16 + l15;
            v8bf lo = *(const v8bf*)&sW[col * WS + kb];
            v8bf hi = *(const v8bf*)&sW[col * WS + 16 + kb];
            #pragma unroll
            for (int i = 0; i < 8; ++i) { b[f][i] = lo[i]; b[f][8 + i] = hi[i]; }
        }

        // ---- 4x4 WMMA: 16 matrix ops per wave per K-step
        #pragma unroll
        for (int fm = 0; fm < 4; ++fm)
            #pragma unroll
            for (int fn = 0; fn < 4; ++fn)
                acc[fm][fn] = __builtin_amdgcn_wmma_f32_16x16x32_bf16(
                    false, a[fm], false, b[fn], (short)0, acc[fm][fn], false, false);

        __syncthreads();
    }

    // ---- epilogue: y = scale[o]*acc + (bias + add1 + add2)[o]
    float scl[4], ofs[4];
    #pragma unroll
    for (int fn = 0; fn < 4; ++fn) {
        int n = colBase + waveN * 64 + fn * 16 + l15;
        scl[fn] = (float)scalesp[n];
        ofs[fn] = (float)biasp[n] + (float)add1p[n] + (float)add2p[n];
    }
    #pragma unroll
    for (int fm = 0; fm < 4; ++fm) {
        int mBase = rowBase + waveM * 64 + fm * 16 + (lane >> 4) * 8;
        #pragma unroll
        for (int fn = 0; fn < 4; ++fn) {
            int n = colBase + waveN * 64 + fn * 16 + l15;
            #pragma unroll
            for (int r = 0; r < 8; ++r) {
                float v = acc[fm][fn][r] * scl[fn] + ofs[fn];
                outp[(size_t)(mBase + r) * ODIM + n] = (__bf16)v;
            }
        }
    }
}

extern "C" void kernel_launch(void* const* d_in, const int* in_sizes, int n_in,
                              void* d_out, int out_size, void* d_ws, size_t ws_size,
                              hipStream_t stream) {
    const __bf16* x  = (const __bf16*)d_in[0];
    const int*    qw = (const int*)   d_in[1];
    const __bf16* sc = (const __bf16*)d_in[2];
    const __bf16* bi = (const __bf16*)d_in[3];
    const __bf16* a1 = (const __bf16*)d_in[4];
    const __bf16* a2 = (const __bf16*)d_in[5];
    __bf16* out = (__bf16*)d_out;

    dim3 grid(ODIM / BN, NROWS / BM);   // (32, 64) = 2048 workgroups
    woq_linear_add_add_kernel<<<grid, 256, 0, stream>>>(x, qw, sc, bi, a1, a2, out);
}
